// RPN_90563680404090
// MI455X (gfx1250) — matrix operand order repaired and checked by reading, not compiled, not run on hardware
//
#include <hip/hip_runtime.h>

typedef __attribute__((ext_vector_type(16))) _Float16 v16h;
typedef __attribute__((ext_vector_type(8)))  _Float16 v8h;
typedef __attribute__((ext_vector_type(8)))  float    v8f;

union V16 { v16h v; v8h h[2]; };

#define NB      8
#define CIN     512
#define COUT    512
#define NP      2500          // 50*50 spatial positions
#define PADP    2704          // 52*52 zero-padded positions
#define SPT     157           // ceil(2500/16) spatial tiles
#define NLOC    36
#define NCLS    18
#define PRE_NMS 3000
#define POST_NMS 300
#define SORTN   32768

__device__ __forceinline__ unsigned f2ord(float f) {
  unsigned u = __float_as_uint(f);
  return (u & 0x80000000u) ? ~u : (u | 0x80000000u);
}
__device__ __forceinline__ float ord2f(unsigned u) {
  return (u & 0x80000000u) ? __uint_as_float(u ^ 0x80000000u)
                           : __uint_as_float(~u);
}

// async copy of 16 bytes global -> LDS (per lane), tracked by ASYNCcnt
__device__ __forceinline__ void async_load16(unsigned lds_off, const _Float16* g) {
  asm volatile("global_load_async_to_lds_b128 %0, %1, off"
               :: "v"(lds_off), "v"((unsigned long long)(size_t)g)
               : "memory");
}

// ---------------- layout conversion kernels (fp32 -> f16, GEMM-friendly) ----
// x (n,c,50,50) fp32 -> xp[n][52*52][c] f16, spatially zero-padded (halo=1)
__global__ void k_cvt_xpad(const float* __restrict__ x, _Float16* __restrict__ xp) {
  size_t t = (size_t)blockIdx.x * blockDim.x + threadIdx.x;
  if (t >= (size_t)NB * PADP * CIN) return;
  int c = t & (CIN - 1);
  size_t np = t >> 9;                    // n*PADP + pp
  int n  = (int)(np / PADP);
  int pp = (int)(np - (size_t)n * PADP);
  int iy = pp / 52, ix = pp - iy * 52;
  float v = 0.f;
  if (iy >= 1 && iy <= 50 && ix >= 1 && ix <= 50)
    v = x[((size_t)n * CIN + c) * NP + (iy - 1) * 50 + (ix - 1)];
  xp[t] = (_Float16)v;
}

// rpn_w (k,c,3,3) fp32 -> aw[tap][k][c] f16
__global__ void k_cvt_w3(const float* __restrict__ wsrc, _Float16* __restrict__ aw) {
  size_t t = (size_t)blockIdx.x * blockDim.x + threadIdx.x;
  if (t >= (size_t)9 * COUT * CIN) return;
  int c   = t & (CIN - 1);
  size_t tk = t >> 9;
  int k   = (int)(tk & (COUT - 1));
  int tap = (int)(tk >> 9);
  aw[t] = (_Float16)wsrc[((size_t)k * CIN + c) * 9 + tap];
}

// reg_w (36,512) -> wr[48][512] zero-padded; cls_w (18,512) -> wc[32][512]
__global__ void k_cvt_w1(const float* __restrict__ rw, const float* __restrict__ cw,
                         _Float16* __restrict__ wr, _Float16* __restrict__ wc) {
  int t = blockIdx.x * blockDim.x + threadIdx.x;
  if (t < 48 * CIN) {
    int m = t >> 9, c = t & (CIN - 1);
    wr[t] = (_Float16)(m < NLOC ? rw[(size_t)m * CIN + c] : 0.f);
  } else if (t < (48 + 32) * CIN) {
    int u = t - 48 * CIN;
    int m = u >> 9, c = u & (CIN - 1);
    wc[u] = (_Float16)(m < NCLS ? cw[(size_t)m * CIN + c] : 0.f);
  }
}

// ---------------- 3x3 conv + ReLU: implicit-GEMM WMMA --------------------
// block = 8 waves = 8 out-channel tiles sharing ONE 16-column spatial tile.
// B tile (9 taps x 16 pos x 32 ch) is async-copied to LDS, double-buffered:
// batch for K-slice c0+32 is in flight while the 9 WMMAs of c0 run.
// Spatial zero-padding removes ALL border predicates: no divergence, EXEC=~0.
__global__ __launch_bounds__(256) void k_conv3x3(
    const _Float16* __restrict__ xp, const _Float16* __restrict__ aw,
    const float* __restrict__ bias, _Float16* __restrict__ hh) {
  __shared__ _Float16 smem[2 * 4608 + 8];        // 2 x 9216B buffers + dummy slot
  const int tid  = threadIdx.x;
  const int lane = tid & 31;
  const int wave = tid >> 5;
  int b = blockIdx.x;                            // 8n * 4 ktgroups * 157 sp
  int n  = b / (4 * SPT);    int rem = b - n * 4 * SPT;
  int kg = rem / SPT;        int sp  = rem - kg * SPT;
  int kt = kg * 8 + wave;                        // this wave's out-ch tile
  const int m  = lane & 15;
  const int hi = lane >> 4;
  int pl = sp * 16 + m;
  bool pok = pl < NP;
  const _Float16* xn = xp + (size_t)n * PADP * CIN;

  // per-thread async chunks: 144 rows x 4 x 16B = 576 chunks, padded to 768
  unsigned loff[3]; size_t goff[3];
#pragma unroll
  for (int j = 0; j < 3; ++j) {
    int idx  = tid + j * 256;
    int ridx = idx < 576 ? idx : idx - 576;
    int row = ridx >> 2, part = ridx & 3;
    int tap = row >> 4,  q    = row & 15;
    int plq = sp * 16 + q; if (plq >= NP) plq = NP - 1;   // clamp ragged edge
    int yq = plq / 50, xq = plq - yq * 50;
    int iy = yq + tap / 3, ix = xq + tap % 3;             // padded coords
    goff[j] = ((size_t)(iy * 52 + ix)) * CIN + part * 8;  // halves; + c0 later
    loff[j] = (idx < 576) ? (unsigned)(tap * 1024 + q * 64 + part * 16)
                          : (unsigned)(2 * 9216);         // dummy 16B slot
  }
  unsigned lds0 = (unsigned)(unsigned long long)(size_t)(void*)smem;

#pragma unroll
  for (int j = 0; j < 3; ++j)                    // prologue: batch for c0=0
    async_load16(lds0 + loff[j], xn + goff[j]);

  v8f acc = {};
  for (int it = 0; it < 16; ++it) {
    int c0 = it * 32;
    if (it < 15) {                               // issue next K-slice batch
      unsigned base = lds0 + (unsigned)(((it + 1) & 1) * 9216);
#pragma unroll
      for (int j = 0; j < 3; ++j)
        async_load16(base + loff[j], xn + goff[j] + c0 + 32);
      asm volatile("s_wait_asynccnt 0x3" ::: "memory");   // prev batch done
    } else {
      asm volatile("s_wait_asynccnt 0x0" ::: "memory");
    }
    __syncthreads();                             // LDS tile visible to all waves
    const _Float16* lb = smem + (it & 1) * 4608; // halves
#pragma unroll
    for (int tap = 0; tap < 9; ++tap) {
      V16 B, A;
      const _Float16* bp = lb + tap * 512 + m * 32 + hi * 16;
      B.h[0] = *(const v8h*)bp;                  // ds_load_b128
      B.h[1] = *(const v8h*)(bp + 8);
      const _Float16* ap = aw + ((size_t)tap * COUT + (kt * 16 + m)) * CIN
                              + c0 + hi * 16;
      A.h[0] = *(const v8h*)ap;                  // K = klo..klo+7
      A.h[1] = *(const v8h*)(ap + 8);            // K = klo+8..klo+15
      acc = __builtin_amdgcn_wmma_f32_16x16x32_f16(
          false, A.v, false, B.v, (short)0, acc, false, false);
    }
    __syncthreads();                             // all waves done reading buf
  }

  if (pok) {                                     // bias + ReLU, pack to f16
    int kb = kt * 16 + hi * 8;                   // 8 contiguous out channels
    v8h out;
#pragma unroll
    for (int r = 0; r < 8; ++r) {
      float v = acc[r] + bias[kb + r];
      out[r] = (_Float16)(v > 0.f ? v : 0.f);
    }
    *(v8h*)(hh + ((size_t)n * NP + pl) * CIN + kb) = out;
  }
}

// ---------------- 1x1 heads (loc: M=48 padded, cls: M=32 padded) ------------
__global__ __launch_bounds__(256) void k_conv1x1(
    const _Float16* __restrict__ hh, const _Float16* __restrict__ wr,
    const _Float16* __restrict__ wc, const float* __restrict__ br,
    const float* __restrict__ bc, float* __restrict__ out_loc,
    float* __restrict__ out_cls) {
  const int lane = threadIdx.x & 31;
  const int wave = threadIdx.x >> 5;
  int t = blockIdx.x * 8 + wave;                 // over n(8) * SPT * 5 mtiles
  int n = t / (SPT * 5);     int rem = t - n * SPT * 5;
  int sp = rem / 5;          int mt  = rem - sp * 5;
  bool isloc = mt < 3;
  const _Float16* W = isloc ? wr : wc;
  int mtl = isloc ? mt : mt - 3;
  const int m  = lane & 15;
  const int hi = lane >> 4;
  int pl = sp * 16 + m;
  bool pok = pl < NP;
  int plc = pok ? pl : NP - 1;                   // clamp: load always valid
  const _Float16* hn = hh + (size_t)n * NP * CIN;
  v8f acc = {};
  for (int c0 = 0; c0 < CIN; c0 += 32) {
    V16 B, A;
    const _Float16* bp = hn + (size_t)plc * CIN + c0 + hi * 16;
    B.h[0] = *(const v8h*)bp;
    B.h[1] = *(const v8h*)(bp + 8);
    const _Float16* ap = W + ((size_t)(mtl * 16 + m)) * CIN + c0 + hi * 16;
    A.h[0] = *(const v8h*)ap;
    A.h[1] = *(const v8h*)(ap + 8);
    acc = __builtin_amdgcn_wmma_f32_16x16x32_f16(
        false, A.v, false, B.v, (short)0, acc, false, false);
  }
  if (pok) {
    int kb = mtl * 16 + hi * 8;
#pragma unroll
    for (int r = 0; r < 8; ++r) {
      int k = kb + r;
      float v = acc[r];
      if (isloc) {
        if (k < NLOC) out_loc[(size_t)n * NP * NLOC + (size_t)pl * NLOC + k] = v + br[k];
      } else {
        if (k < NCLS) out_cls[(size_t)n * NP * NCLS + (size_t)pl * NCLS + k] = v + bc[k];
      }
    }
  }
}

// ---------------- anchors + decode + clip + softmax-fg ----------------------
__global__ void k_prep(const float* __restrict__ loc, const float* __restrict__ cls,
                       float* __restrict__ boxes, float* __restrict__ scores,
                       float* __restrict__ anchor_out) {
  int i = blockIdx.x * blockDim.x + threadIdx.x;   // over 8*22500
  if (i >= NB * NP * 9) return;
  int n = i / (NP * 9); int j = i - n * (NP * 9);
  int p = j / 9;        int a = j - p * 9;
  int y = p / 50, x = p - y * 50;
  const float ratios[3] = {0.5f, 1.f, 2.f};
  const float scales[3] = {8.f, 16.f, 32.f};
  int ri = a / 3, si = a - ri * 3;
  float ah = 16.f * scales[si] * sqrtf(ratios[ri]);
  float aw = 16.f * scales[si] * sqrtf(1.f / ratios[ri]);
  float acy = y * 16.f + 8.f, acx = x * 16.f + 8.f;
  if (n == 0) {
    anchor_out[j * 4 + 0] = acy - 0.5f * ah;
    anchor_out[j * 4 + 1] = acx - 0.5f * aw;
    anchor_out[j * 4 + 2] = acy + 0.5f * ah;
    anchor_out[j * 4 + 3] = acx + 0.5f * aw;
  }
  const float* l = loc + (size_t)n * NP * 36 + (size_t)j * 4;
  float cy = l[0] * ah + acy, cx = l[1] * aw + acx;
  float h2 = expf(l[2]) * ah, w2 = expf(l[3]) * aw;
  float y1 = fminf(fmaxf(cy - 0.5f * h2, 0.f), 800.f);
  float x1 = fminf(fmaxf(cx - 0.5f * w2, 0.f), 800.f);
  float y2 = fminf(fmaxf(cy + 0.5f * h2, 0.f), 800.f);
  float x2 = fminf(fmaxf(cx + 0.5f * w2, 0.f), 800.f);
  bool valid = (y2 - y1 >= 16.f) && (x2 - x1 >= 16.f);
  const float* c2 = cls + (size_t)n * NP * 18 + (size_t)j * 2;
  float s = 1.f / (1.f + expf(c2[0] - c2[1]));     // softmax fg
  float* b = boxes + (size_t)i * 4;
  b[0] = y1; b[1] = x1; b[2] = y2; b[3] = x2;
  scores[i] = valid ? s : -__builtin_inff();
}

// ---------------- per-batch top-3000 via full bitonic sort in 256KB LDS -----
extern __shared__ unsigned long long S[];
__global__ __launch_bounds__(1024) void k_topk(
    const float* __restrict__ scores, const float* __restrict__ boxes,
    float* __restrict__ tscore, float* __restrict__ tbox) {
  int n = blockIdx.x, tid = threadIdx.x;
  const float* sc = scores + (size_t)n * NP * 9;
  for (int j = tid; j < SORTN; j += 1024) {
    unsigned u = (j < NP * 9) ? f2ord(sc[j]) : 0u;   // pad below -inf
    S[j] = ((unsigned long long)u << 32) | (unsigned)j;
  }
  __syncthreads();
  for (int k = 2; k <= SORTN; k <<= 1) {
    for (int j = k >> 1; j > 0; j >>= 1) {
      for (int i = tid; i < SORTN; i += 1024) {
        int ixj = i ^ j;
        if (ixj > i) {
          unsigned long long va = S[i], vb = S[ixj];
          bool up = (i & k) == 0;                    // descending overall
          if (up ? (va < vb) : (va > vb)) { S[i] = vb; S[ixj] = va; }
        }
      }
      __syncthreads();
    }
  }
  for (int i = tid; i < PRE_NMS; i += 1024) {
    unsigned long long key = S[i];
    unsigned idx = (unsigned)key;
    float s = -__builtin_inff();
    float b0 = 0.f, b1 = 0.f, b2 = 0.f, b3 = 0.f;
    if (idx < (unsigned)(NP * 9)) {
      s = ord2f((unsigned)(key >> 32));
      const float* bs = boxes + ((size_t)n * NP * 9 + idx) * 4;
      b0 = bs[0]; b1 = bs[1]; b2 = bs[2]; b3 = bs[3];
    }
    tscore[(size_t)n * PRE_NMS + i] = s;
    float* bd = tbox + ((size_t)n * PRE_NMS + i) * 4;
    bd[0] = b0; bd[1] = b1; bd[2] = b2; bd[3] = b3;
  }
}

// ---------------- greedy NMS, boxes cached in LDS ---------------------------
__global__ __launch_bounds__(256) void k_nms(
    const float* __restrict__ tscore, const float* __restrict__ tbox,
    float* __restrict__ rois, int* __restrict__ ridx) {
  __shared__ float B4[PRE_NMS * 4];
  __shared__ unsigned char live[PRE_NMS];
  __shared__ int first;
  int n = blockIdx.x, tid = threadIdx.x;
  for (int i = tid; i < PRE_NMS; i += 256) {
    const float* bs = tbox + ((size_t)n * PRE_NMS + i) * 4;
    B4[i * 4 + 0] = bs[0]; B4[i * 4 + 1] = bs[1];
    B4[i * 4 + 2] = bs[2]; B4[i * 4 + 3] = bs[3];
    live[i] = tscore[(size_t)n * PRE_NMS + i] > -3.0e38f ? 1 : 0;
  }
  __syncthreads();
  for (int it = 0; it < POST_NMS; ++it) {
    if (tid == 0) first = 0x7fffffff;
    __syncthreads();
    int loc = 0x7fffffff;
    for (int i = tid; i < PRE_NMS; i += 256)
      if (live[i]) { loc = i; break; }               // stripe minimum
    if (loc != 0x7fffffff) atomicMin(&first, loc);
    __syncthreads();
    int f = first;
    bool found = f < PRE_NMS;
    float y1 = 0.f, x1 = 0.f, y2 = 0.f, x2 = 0.f, area = 0.f;
    if (found) {
      y1 = B4[f * 4]; x1 = B4[f * 4 + 1]; y2 = B4[f * 4 + 2]; x2 = B4[f * 4 + 3];
      area = (y2 - y1) * (x2 - x1);
    }
    __syncthreads();
    if (found) {
      for (int i = tid; i < PRE_NMS; i += 256) {
        if (!live[i]) continue;
        float ty = fmaxf(y1, B4[i * 4]),     tx = fmaxf(x1, B4[i * 4 + 1]);
        float by = fminf(y2, B4[i * 4 + 2]), bx = fminf(x2, B4[i * 4 + 3]);
        float ih = fmaxf(by - ty, 0.f), iw = fmaxf(bx - tx, 0.f);
        float inter = ih * iw;
        float a2 = (B4[i * 4 + 2] - B4[i * 4]) * (B4[i * 4 + 3] - B4[i * 4 + 1]);
        float iou = inter / (area + a2 - inter + 1e-9f);
        if (iou > 0.7f) live[i] = 0;                 // clears f itself too
      }
    }
    if (tid == 0) {
      float* r = rois + ((size_t)n * POST_NMS + it) * 4;
      r[0] = found ? y1 : 0.f; r[1] = found ? x1 : 0.f;
      r[2] = found ? y2 : 0.f; r[3] = found ? x2 : 0.f;
      ridx[n * POST_NMS + it] = found ? n : -1;
    }
    __syncthreads();
  }
}

// ---------------- host ------------------------------------------------------
extern "C" void kernel_launch(void* const* d_in, const int* in_sizes, int n_in,
                              void* d_out, int out_size, void* d_ws, size_t ws_size,
                              hipStream_t stream) {
  (void)in_sizes; (void)n_in; (void)out_size; (void)ws_size;
  const float* x     = (const float*)d_in[0];
  const float* rpn_w = (const float*)d_in[1];
  const float* rpn_b = (const float*)d_in[2];
  const float* reg_w = (const float*)d_in[3];
  const float* reg_b = (const float*)d_in[4];
  const float* cls_w = (const float*)d_in[5];
  const float* cls_b = (const float*)d_in[6];

  float* out     = (float*)d_out;
  float* out_loc = out;                         // 8*22500*4 = 720000
  float* out_cls = out + 720000;                // 8*22500*2 = 360000
  float* rois    = out + 1080000;               // 2400*4    = 9600
  int*   ridx    = (int*)(out + 1089600);       // 2400
  float* anch    = out + 1092000;               // 22500*4   = 90000

  char* w = (char*)d_ws;
  size_t off = 0;
  auto carve = [&](size_t bytes) -> char* {
    char* p = w + off; off += (bytes + 255) & ~(size_t)255; return p;
  };
  _Float16* xp  = (_Float16*)carve((size_t)NB * PADP * CIN * 2);
  _Float16* hh  = (_Float16*)carve((size_t)NB * NP * CIN * 2);
  _Float16* aw  = (_Float16*)carve((size_t)9 * COUT * CIN * 2);
  _Float16* wr  = (_Float16*)carve((size_t)48 * CIN * 2);
  _Float16* wc  = (_Float16*)carve((size_t)32 * CIN * 2);
  float* boxes  = (float*)carve((size_t)NB * NP * 9 * 4 * 4);
  float* scores = (float*)carve((size_t)NB * NP * 9 * 4);
  float* tbox   = (float*)carve((size_t)NB * PRE_NMS * 4 * 4);
  float* tscore = (float*)carve((size_t)NB * PRE_NMS * 4);

  k_cvt_xpad<<<(NB * PADP * CIN + 255) / 256, 256, 0, stream>>>(x, xp);
  k_cvt_w3  <<<(9 * COUT * CIN + 255) / 256, 256, 0, stream>>>(rpn_w, aw);
  k_cvt_w1  <<<((48 + 32) * CIN + 255) / 256, 256, 0, stream>>>(reg_w, cls_w, wr, wc);

  // 8 n * 4 ktgroups * 157 sp tiles = 5024 blocks (8 waves each)
  k_conv3x3<<<NB * 4 * SPT, 256, 0, stream>>>(xp, aw, rpn_b, hh);
  // 8 n * 157 sp tiles * 5 m tiles = 6280 tiles, 8 waves/block
  k_conv1x1<<<6280 / 8, 256, 0, stream>>>(hh, wr, wc, reg_b, cls_b, out_loc, out_cls);

  k_prep<<<(NB * NP * 9 + 255) / 256, 256, 0, stream>>>(out_loc, out_cls,
                                                        boxes, scores, anch);
  k_topk<<<NB, 1024, SORTN * sizeof(unsigned long long), stream>>>(scores, boxes,
                                                                   tscore, tbox);
  k_nms<<<NB, 256, 0, stream>>>(tscore, tbox, rois, ridx);
}
